// WeightedMMDLoss_10651518894538
// MI455X (gfx1250) — compile-verified, matrix operand order checked by reading, not run on hardware
//
#include <hip/hip_runtime.h>
#include <hip/hip_bf16.h>

#define NROW  4096
#define DIM   256
#define TOT   8192
#define TILE  128
#define KSTEP 32
#define KITER (DIM / KSTEP)

typedef __attribute__((ext_vector_type(16))) __bf16        v16bf;
typedef __attribute__((ext_vector_type(8)))  float         v8f;
typedef __attribute__((ext_vector_type(4)))  unsigned int  u32x4;

union Frag { v16bf v; u32x4 q[2]; };

__device__ __forceinline__ unsigned short f32_to_bf16(float f) {
    unsigned int u = __float_as_uint(f);
    u += 0x7FFFu + ((u >> 16) & 1u);          // round-to-nearest-even
    return (unsigned short)(u >> 16);
}
__device__ __forceinline__ float bf16_to_f32(unsigned short h) {
    return __uint_as_float(((unsigned int)h) << 16);
}

// Async DMA: 16 bytes global -> LDS per lane, tracked by ASYNCcnt (no VGPR data).
__device__ __forceinline__ void async_ld16(const void* g, const void* lds) {
    unsigned off = (unsigned)(size_t)lds;     // LDS offset = low 32 bits of generic address
    asm volatile("global_load_async_to_lds_b128 %0, %1, off"
                 :: "v"(off), "v"(g) : "memory");
}

// ---- ws header: hdr[0]=sumL2, hdr[1..4]=quadrant sums (XX,XY,YX,YY), hdr[5]=sumW ----

__global__ void k_init(double* hdr) {
    if (threadIdx.x < 6) hdr[threadIdx.x] = 0.0;
}

// Row prep: sq[i] = ||row||^2 ; split row into bf16 hi + bf16 lo.
__global__ __launch_bounds__(256) void k_prep(const float* __restrict__ src,
                                              const float* __restrict__ tgt,
                                              float* __restrict__ sq,
                                              unsigned short* __restrict__ ahi,
                                              unsigned short* __restrict__ alo) {
    int warp = threadIdx.x >> 5, lane = threadIdx.x & 31;
    int row  = blockIdx.x * 8 + warp;
    const float* p = (row < NROW) ? (src + (size_t)row * DIM)
                                  : (tgt + (size_t)(row - NROW) * DIM);
    float s = 0.f;
    #pragma unroll
    for (int j = 0; j < DIM; j += 32) {
        float x = p[j + lane];
        s += x * x;
        unsigned short hi = f32_to_bf16(x);
        float lof = x - bf16_to_f32(hi);
        size_t idx = (size_t)row * DIM + j + lane;
        ahi[idx] = hi;
        alo[idx] = f32_to_bf16(lof);
    }
    #pragma unroll
    for (int off = 16; off; off >>= 1) s += __shfl_down(s, off, 32);
    if (lane == 0) sq[row] = s;
}

// sumW = sum(sigmoid(|s1-s2|))  (W weighting collapses to (sumW)^2 since XX is scalar)
__global__ __launch_bounds__(256) void k_scores(const float* __restrict__ s1,
                                                const float* __restrict__ s2,
                                                double* hdr) {
    __shared__ float red[8];
    int tid = threadIdx.x, lane = tid & 31, warp = tid >> 5;
    int i = blockIdx.x * 256 + tid;
    float d = fabsf(s1[i] - s2[i]);
    float w = 1.0f / (1.0f + __builtin_amdgcn_exp2f(-d * 1.44269504f));
    #pragma unroll
    for (int off = 16; off; off >>= 1) w += __shfl_down(w, off, 32);
    if (lane == 0) red[warp] = w;
    __syncthreads();
    if (tid == 0) {
        float t = 0.f;
        #pragma unroll
        for (int k = 0; k < 8; k++) t += red[k];
        atomicAdd(&hdr[5], (double)t);
    }
}

// Fused GEMM (bf16 hi/lo split WMMA, async double-buffered LDS) + distance epilogue.
// pass==1: accumulate sum(max(L2,0)) -> hdr[0]
// pass==2: accumulate 5-bandwidth RBF kernel sums into quadrant accumulators hdr[1..4]
__global__ __launch_bounds__(256) void k_mmd_gemm(const unsigned short* __restrict__ ahi,
                                                  const unsigned short* __restrict__ alo,
                                                  const float* __restrict__ sq,
                                                  double* hdr, int pass) {
    int tileJ = blockIdx.x, tileI = blockIdx.y;
    if (tileI > tileJ) return;                      // symmetric: upper triangle only

    __shared__ unsigned short sAhi[2][TILE * KSTEP], sAlo[2][TILE * KSTEP];
    __shared__ unsigned short sBhi[2][TILE * KSTEP], sBlo[2][TILE * KSTEP];
    __shared__ float sqI[TILE], sqJ[TILE];
    __shared__ float red[8];

    int tid  = threadIdx.x;
    int lane = tid & 31, warp = tid >> 5;
    int wm = warp & 3, wn = warp >> 2;              // 4x2 wave grid over 128x128 tile
    int rowI = tileI * TILE, rowJ = tileJ * TILE;

    if (tid < TILE) sqI[tid]        = sq[rowI + tid];
    else            sqJ[tid - TILE] = sq[rowJ + (tid - TILE)];

    // Issue one stage batch: 8 async b128 DMAs per wave (2 chunks x 4 planes).
    auto stage = [&](int buf, int kk) {
        #pragma unroll
        for (int c = tid; c < TILE * 4; c += 256) {
            int r   = c >> 2;
            int off = (c & 3) << 3;                 // 8 shorts = 16 bytes
            size_t gI = (size_t)(rowI + r) * DIM + kk + off;
            size_t gJ = (size_t)(rowJ + r) * DIM + kk + off;
            int l = r * KSTEP + off;
            async_ld16(&ahi[gI], &sAhi[buf][l]);
            async_ld16(&alo[gI], &sAlo[buf][l]);
            async_ld16(&ahi[gJ], &sBhi[buf][l]);
            async_ld16(&alo[gJ], &sBlo[buf][l]);
        }
    };

    v8f C[2][4];
    #pragma unroll
    for (int s = 0; s < 2; s++)
        #pragma unroll
        for (int t = 0; t < 4; t++)
            C[s][t] = (v8f){0.f, 0.f, 0.f, 0.f, 0.f, 0.f, 0.f, 0.f};

    stage(0, 0);                                    // prologue prefetch

    for (int ki = 0; ki < KITER; ++ki) {
        int buf = ki & 1;
        if (ki + 1 < KITER) {
            stage(buf ^ 1, (ki + 1) * KSTEP);       // prefetch next k-step
            // async loads complete in order: <=8 outstanding -> current batch landed
            asm volatile("s_wait_asynccnt 0x8" ::: "memory");
        } else {
            asm volatile("s_wait_asynccnt 0x0" ::: "memory");
        }
        __syncthreads();                            // all waves' DMA visible

        // bf16 16x32 fragment layout: lanes 0-15 hold K[0..7],K[16..23];
        // lanes 16-31 hold K[8..15],K[24..31] -> two b128 LDS loads per fragment.
        int ko = (lane < 16) ? 0 : 8;
        Frag aHi[2], aLo[2], bHi[4], bLo[4];
        #pragma unroll
        for (int s = 0; s < 2; s++) {
            int ro = (wm * 32 + s * 16 + (lane & 15)) * KSTEP;
            aHi[s].q[0] = *(const u32x4*)&sAhi[buf][ro + ko];
            aHi[s].q[1] = *(const u32x4*)&sAhi[buf][ro + ko + 16];
            aLo[s].q[0] = *(const u32x4*)&sAlo[buf][ro + ko];
            aLo[s].q[1] = *(const u32x4*)&sAlo[buf][ro + ko + 16];
        }
        #pragma unroll
        for (int t = 0; t < 4; t++) {
            int ro = (wn * 64 + t * 16 + (lane & 15)) * KSTEP;
            bHi[t].q[0] = *(const u32x4*)&sBhi[buf][ro + ko];
            bHi[t].q[1] = *(const u32x4*)&sBhi[buf][ro + ko + 16];
            bLo[t].q[0] = *(const u32x4*)&sBlo[buf][ro + ko];
            bLo[t].q[1] = *(const u32x4*)&sBlo[buf][ro + ko + 16];
        }

        // G ~= hi*hi + hi*lo + lo*hi  (near-fp32 accuracy at bf16 matrix rate)
        #pragma unroll
        for (int s = 0; s < 2; s++)
            #pragma unroll
            for (int t = 0; t < 4; t++) {
                C[s][t] = __builtin_amdgcn_wmma_f32_16x16x32_bf16(
                    false, aHi[s].v, false, bHi[t].v, (short)0, C[s][t], false, false);
                C[s][t] = __builtin_amdgcn_wmma_f32_16x16x32_bf16(
                    false, aHi[s].v, false, bLo[t].v, (short)0, C[s][t], false, false);
                C[s][t] = __builtin_amdgcn_wmma_f32_16x16x32_bf16(
                    false, aLo[s].v, false, bHi[t].v, (short)0, C[s][t], false, false);
            }
        __syncthreads();                            // done reading buf before overwrite
    }

    float c0 = 0.f;
    if (pass == 2) {
        double nd = (double)TOT;
        double bw = hdr[0] / (nd * nd - nd) / 4.0;   // / KERNEL_MUL^(KERNEL_NUM/2)
        c0 = (float)(1.4426950408889634 / bw);       // log2(e)/bw
    }

    // Epilogue: L2 = sq[m]+sq[n]-2G ; per C/D layout lane L: n=L&15, m=r+8*(L>>4)
    float acc = 0.f;
    int halfLane = lane >> 4;
    int nl = wn * 64 + (lane & 15);
    #pragma unroll
    for (int s = 0; s < 2; s++) {
        int mbase = wm * 32 + s * 16 + halfLane * 8;
        #pragma unroll
        for (int t = 0; t < 4; t++) {
            float sqn = sqJ[nl + t * 16];
            #pragma unroll
            for (int r = 0; r < 8; r++) {
                float l2 = fmaxf(sqI[mbase + r] + sqn - 2.0f * C[s][t][r], 0.f);
                if (pass == 1) {
                    acc += l2;
                } else {
                    float e = -l2 * c0;              // exp(-L2/(bw*2^i)) = 2^(e*2^-i)
                    acc += __builtin_amdgcn_exp2f(e)
                         + __builtin_amdgcn_exp2f(e * 0.5f)
                         + __builtin_amdgcn_exp2f(e * 0.25f)
                         + __builtin_amdgcn_exp2f(e * 0.125f)
                         + __builtin_amdgcn_exp2f(e * 0.0625f);
                }
            }
        }
    }
    #pragma unroll
    for (int off = 16; off; off >>= 1) acc += __shfl_down(acc, off, 32);
    if (lane == 0) red[warp] = acc;
    __syncthreads();
    if (tid == 0) {
        float tsum = 0.f;
        #pragma unroll
        for (int w2 = 0; w2 < 8; w2++) tsum += red[w2];
        if (pass == 1) {
            atomicAdd(&hdr[0], (double)tsum * ((tileI == tileJ) ? 1.0 : 2.0));
        } else {
            int qi = (rowI >= NROW), qj = (rowJ >= NROW);
            atomicAdd(&hdr[1 + qi * 2 + qj], (double)tsum);
            if (tileI != tileJ) atomicAdd(&hdr[1 + qj * 2 + qi], (double)tsum);
        }
    }
}

__global__ void k_final(const double* hdr, float* out) {
    double b   = (double)NROW;
    double inv = 1.0 / (b * b);
    double XX = hdr[1] * inv, XY = hdr[2] * inv, YX = hdr[3] * inv, YY = hdr[4] * inv;
    double sw = hdr[5];
    double S  = sw * sw;
    double XXw = XX * (S / (S + 1e-8));
    out[0] = (float)(XXw + YY - XY - YX);
}

extern "C" void kernel_launch(void* const* d_in, const int* in_sizes, int n_in,
                              void* d_out, int out_size, void* d_ws, size_t ws_size,
                              hipStream_t stream) {
    (void)in_sizes; (void)n_in; (void)out_size; (void)ws_size;
    const float* src = (const float*)d_in[0];
    const float* tgt = (const float*)d_in[1];
    const float* s1  = (const float*)d_in[2];
    const float* s2  = (const float*)d_in[3];

    char* ws = (char*)d_ws;
    double*         hdr = (double*)ws;                               // 64 B header
    float*          sq  = (float*)(ws + 64);                        // 8192 f32
    unsigned short* ahi = (unsigned short*)(ws + 64 + TOT * 4);     // 8192x256 bf16
    unsigned short* alo = (unsigned short*)(ws + 64 + TOT * 4 + (size_t)TOT * DIM * 2);

    k_init  <<<1, 32, 0, stream>>>(hdr);
    k_prep  <<<TOT / 8, 256, 0, stream>>>(src, tgt, sq, ahi, alo);
    k_scores<<<NROW / 256, 256, 0, stream>>>(s1, s2, hdr);

    dim3 grid(TOT / TILE, TOT / TILE);
    k_mmd_gemm<<<grid, 256, 0, stream>>>(ahi, alo, sq, hdr, 1);
    k_mmd_gemm<<<grid, 256, 0, stream>>>(ahi, alo, sq, hdr, 2);
    k_final <<<1, 1, 0, stream>>>(hdr, (float*)d_out);
}
